// FastWeightMemory_73280732005022
// MI455X (gfx1250) — matrix operand order repaired and checked by reading, not compiled, not run on hardware
//
#include <hip/hip_runtime.h>

typedef __attribute__((ext_vector_type(16))) __bf16 v16bf;
typedef __attribute__((ext_vector_type(8)))  __bf16 v8bf;
typedef __attribute__((ext_vector_type(8)))  float  v8f;

#define B_      4
#define S_      4096
#define H_      1024
#define MEM_    256
#define CHUNK_  64      // time steps per chunk
#define NCHUNK_ 64
#define ROWS_   (S_ * B_)   // 16384 (row = s*B + b, time-major)
#define DECAY_  0.99f
#define PAD_    40          // LDS stride (elems) for 32-wide K tiles
#define SPAD_   264         // LDS stride for 256-wide score tile

// ---------- CDNA5 async global->LDS path (compile-safe detection) ----------
#if __has_builtin(__builtin_amdgcn_global_load_async_to_lds_b128) && \
    __has_builtin(__builtin_amdgcn_s_wait_asynccnt)
#define FWM_ASYNC 1
#else
#define FWM_ASYNC 0
#endif

#if FWM_ASYNC
// The builtin expects gcc-style int4 pointers: AS1 (global) source, AS3 (LDS) dest.
// (clang prints AS1 as "__device__" in HIP mode — confirmed by diagnostics.)
typedef int v4i_fwm __attribute__((vector_size(16)));
typedef v4i_fwm __attribute__((address_space(1)))* fwm_gp_t;
typedef v4i_fwm __attribute__((address_space(3)))* fwm_lp_t;

// 16-byte per-lane async copy, global -> LDS, tracked by ASYNCcnt.
// Generic->AS casts done through integers: low 32 bits of a generic LDS
// pointer are the LDS byte offset (aperture rules, ISA 10.2).
static __device__ __forceinline__ void async_cp16(const void* g, void* l) {
  __builtin_amdgcn_global_load_async_to_lds_b128(
      (fwm_gp_t)(unsigned long long)g,
      (fwm_lp_t)(unsigned int)(unsigned long long)l,
      0, 0);
}
#define FWM_WAIT_ASYNC() __builtin_amdgcn_s_wait_asynccnt(0)
#else
#define FWM_WAIT_ASYNC()
#endif

// ---------- WMMA fragment helpers (exact gfx1250 VGPR layouts) ----------

static __device__ __forceinline__ v8f vzero8() {
  v8f z;
#pragma unroll
  for (int i = 0; i < 8; ++i) z[i] = 0.0f;
  return z;
}

// A 16x32 bf16: lane<16 -> M=lane, K in {0..7, 16..23}; lane>=16 -> M=lane-16, K in {8..15, 24..31}
static __device__ __forceinline__ v16bf load_a_frag(const __bf16* base, int stride) {
  int lane = threadIdx.x & 31;
  int m  = lane & 15;
  int kb = (lane >> 4) << 3;   // 0 or 8
  const __bf16* p = base + m * stride + kb;
  v8bf lo = *(const v8bf*)(p);
  v8bf hi = *(const v8bf*)(p + 16);
  v16bf f;
#pragma unroll
  for (int i = 0; i < 8; ++i) { f[i] = lo[i]; f[i + 8] = hi[i]; }
  return f;
}

// B 32x16 bf16 from N-major tile Bt[n][k]: lane<16 -> N=lane, K in {0..7, 16..23}; lane>=16 -> N=lane-16, K in {8..15, 24..31}
static __device__ __forceinline__ v16bf load_b_frag(const __bf16* base, int stride) {
  int lane = threadIdx.x & 31;
  int n  = lane & 15;
  int kb = (lane >> 4) << 3;   // 0 or 8
  const __bf16* p = base + n * stride + kb;
  v8bf lo = *(const v8bf*)(p);
  v8bf hi = *(const v8bf*)(p + 16);
  v16bf f;
#pragma unroll
  for (int i = 0; i < 8; ++i) { f[i] = lo[i]; f[i + 8] = hi[i]; }
  return f;
}

#define WMMA_BF16(Af, Bf, Cf) \
  __builtin_amdgcn_wmma_f32_16x16x32_bf16(false, (Af), false, (Bf), (short)0, (Cf), false, false)

// ---------- kernel: convert + reorder x (b,s,h) f32 -> xr[(s*B+b)][h] bf16 ----------
__global__ void fwm_convert_x(const float* __restrict__ x, __bf16* __restrict__ xr) {
  int idx = (blockIdx.x * 256 + threadIdx.x) * 8;   // flat over (b,s,h)
  int b   = idx >> 22;                              // / (4096*1024)
  int rem = idx & ((1 << 22) - 1);
  int s   = rem >> 10;
  int h   = rem & 1023;
  const float4* p = (const float4*)(x + idx);
  float4 lo = p[0], hi = p[1];
  v8bf o;
  o[0] = (__bf16)lo.x; o[1] = (__bf16)lo.y; o[2] = (__bf16)lo.z; o[3] = (__bf16)lo.w;
  o[4] = (__bf16)hi.x; o[5] = (__bf16)hi.y; o[6] = (__bf16)hi.z; o[7] = (__bf16)hi.w;
  *(v8bf*)(xr + (s * B_ + b) * H_ + h) = o;
}

// ---------- kernel: f32 -> bf16 bulk convert (weights) ----------
__global__ void fwm_convert_w(const float* __restrict__ in, __bf16* __restrict__ out) {
  int i = (blockIdx.x * 256 + threadIdx.x) * 8;
  const float4* p = (const float4*)(in + i);
  float4 lo = p[0], hi = p[1];
  v8bf o;
  o[0] = (__bf16)lo.x; o[1] = (__bf16)lo.y; o[2] = (__bf16)lo.z; o[3] = (__bf16)lo.w;
  o[4] = (__bf16)hi.x; o[5] = (__bf16)hi.y; o[6] = (__bf16)hi.z; o[7] = (__bf16)hi.w;
  *(v8bf*)(out + i) = o;
}

// ---------- kernel: generic bf16 WMMA GEMM  C[M,N] = A[M,K] * Bt[N,K]^T ----------
// block tile 128x128, 8 waves (2x4), wave tile 64x32
// Double-buffered async global->LDS pipeline when available.
// Cbf != null : bf16 row-major output (ldc)
// Cfin != null: final mode — scatter row (s*B+b) -> (b,s,:) into f32 out, * sigmoid(gate)
__global__ void __launch_bounds__(256) fwm_gemm(
    const __bf16* __restrict__ A, int lda,
    const __bf16* __restrict__ Bt, int ldb,
    int K,
    __bf16* __restrict__ Cbf, int ldc,
    float* __restrict__ Cfin, const float* __restrict__ gate) {
  __shared__ __align__(16) __bf16 Als[2][128 * PAD_];
  __shared__ __align__(16) __bf16 Bls[2][128 * PAD_];
  const int tid  = threadIdx.x;
  const int lane = tid & 31;
  const int wid  = tid >> 5;
  const int wm   = wid >> 2;            // 0..1
  const int wn   = wid & 3;             // 0..3
  const int blockM = blockIdx.x * 128;
  const int blockN = blockIdx.y * 128;

  v8f acc[4][2];
#pragma unroll
  for (int mi = 0; mi < 4; ++mi)
#pragma unroll
    for (int ni = 0; ni < 2; ++ni) acc[mi][ni] = vzero8();

  const int srow  = tid >> 1;           // 0..127
  const int shalf = (tid & 1) * 16;

#if FWM_ASYNC
  auto issue = [&](int k0, int buf) {
    const __bf16* ga = A  + (blockM + srow) * lda + k0 + shalf;
    const __bf16* gb = Bt + (blockN + srow) * ldb + k0 + shalf;
    __bf16* la = Als[buf] + srow * PAD_ + shalf;
    __bf16* lb = Bls[buf] + srow * PAD_ + shalf;
    async_cp16(ga,     la);
    async_cp16(ga + 8, la + 8);
    async_cp16(gb,     lb);
    async_cp16(gb + 8, lb + 8);
  };
  issue(0, 0);
  int buf = 0;
  for (int k0 = 0; k0 < K; k0 += 32, buf ^= 1) {
    FWM_WAIT_ASYNC();                   // this wave's copies for tile `buf` done
    __syncthreads();                    // everyone's copies done + prev reads drained
    if (k0 + 32 < K) issue(k0 + 32, buf ^ 1);
    v16bf afrag[4];
#pragma unroll
    for (int mi = 0; mi < 4; ++mi)
      afrag[mi] = load_a_frag(Als[buf] + (wm * 64 + mi * 16) * PAD_, PAD_);
#pragma unroll
    for (int ni = 0; ni < 2; ++ni) {
      v16bf bfrag = load_b_frag(Bls[buf] + (wn * 32 + ni * 16) * PAD_, PAD_);
#pragma unroll
      for (int mi = 0; mi < 4; ++mi)
        acc[mi][ni] = WMMA_BF16(afrag[mi], bfrag, acc[mi][ni]);
    }
  }
  __syncthreads();
#else
  for (int k0 = 0; k0 < K; k0 += 32) {
    const __bf16* ga = A  + (blockM + srow) * lda + k0 + shalf;
    const __bf16* gb = Bt + (blockN + srow) * ldb + k0 + shalf;
    v8bf a0 = *(const v8bf*)ga, a1 = *(const v8bf*)(ga + 8);
    v8bf b0 = *(const v8bf*)gb, b1 = *(const v8bf*)(gb + 8);
    if (k0 + 32 < K) {
      __builtin_prefetch(ga + 32, 0, 3);
      __builtin_prefetch(gb + 32, 0, 3);
    }
    *(v8bf*)(Als[0] + srow * PAD_ + shalf)     = a0;
    *(v8bf*)(Als[0] + srow * PAD_ + shalf + 8) = a1;
    *(v8bf*)(Bls[0] + srow * PAD_ + shalf)     = b0;
    *(v8bf*)(Bls[0] + srow * PAD_ + shalf + 8) = b1;
    __syncthreads();
    v16bf afrag[4];
#pragma unroll
    for (int mi = 0; mi < 4; ++mi)
      afrag[mi] = load_a_frag(Als[0] + (wm * 64 + mi * 16) * PAD_, PAD_);
#pragma unroll
    for (int ni = 0; ni < 2; ++ni) {
      v16bf bfrag = load_b_frag(Bls[0] + (wn * 32 + ni * 16) * PAD_, PAD_);
#pragma unroll
      for (int mi = 0; mi < 4; ++mi)
        acc[mi][ni] = WMMA_BF16(afrag[mi], bfrag, acc[mi][ni]);
    }
    __syncthreads();
  }
#endif

  const int crow = (lane >> 4) * 8;     // 0 or 8
  const int ccol = lane & 15;
  if (Cfin) {
    const float g = 1.0f / (1.0f + expf(-gate[0]));
#pragma unroll
    for (int mi = 0; mi < 4; ++mi)
#pragma unroll
      for (int ni = 0; ni < 2; ++ni)
#pragma unroll
        for (int v = 0; v < 8; ++v) {
          int gm = blockM + wm * 64 + mi * 16 + crow + v;
          int gn = blockN + wn * 32 + ni * 16 + ccol;
          int b = gm & 3, s = gm >> 2;
          Cfin[b * (S_ * H_) + s * H_ + gn] = acc[mi][ni][v] * g;
        }
  } else {
#pragma unroll
    for (int mi = 0; mi < 4; ++mi)
#pragma unroll
      for (int ni = 0; ni < 2; ++ni)
#pragma unroll
        for (int v = 0; v < 8; ++v) {
          int gm = blockM + wm * 64 + mi * 16 + crow + v;
          int gn = blockN + wn * 32 + ni * 16 + ccol;
          Cbf[gm * ldc + gn] = (__bf16)acc[mi][ni][v];
        }
  }
}

// ---------- kernel: per-chunk decay-weighted outer-product sums ----------
// G_c[m][n] = sum_rows decay^(63 - tau(row)) * v[row][m] * k[row][n] / B
// grid (NCHUNK_, 2), block 256; block tile 128(m) x 256(n), K = 256 chunk rows
__global__ void __launch_bounds__(256) fwm_chunk_sums(
    const __bf16* __restrict__ vmat, const __bf16* __restrict__ kmat,
    float* __restrict__ G) {
  __shared__ __align__(16) __bf16 Als[128 * PAD_];   // (w*V)^T tile: [m][kr]
  __shared__ __align__(16) __bf16 Bls[256 * PAD_];   // K^T tile:     [n][kr]
  const int tid  = threadIdx.x;
  const int lane = tid & 31;
  const int wid  = tid >> 5;
  const int wm   = wid >> 2;
  const int wn   = wid & 3;
  const int chunk = blockIdx.x;
  const int m0    = blockIdx.y * 128;
  const int rowbase = chunk * 256;

  v8f acc[4][4];
#pragma unroll
  for (int mi = 0; mi < 4; ++mi)
#pragma unroll
    for (int ni = 0; ni < 4; ++ni) acc[mi][ni] = vzero8();

  const int kr  = tid >> 3;  // 0..31
  const int seg = tid & 7;   // 0..7

  for (int k0 = 0; k0 < 256; k0 += 32) {
    const int grow = rowbase + k0 + kr;
    const float w = 0.25f * powf(DECAY_, (float)(CHUNK_ - 1 - ((k0 + kr) >> 2)));
    {   // A: transposed weighted V, 16 m per thread
      const v8bf* p = (const v8bf*)(vmat + grow * MEM_ + m0 + seg * 16);
      v8bf x0 = p[0], x1 = p[1];
#pragma unroll
      for (int e = 0; e < 8; ++e) {
        Als[(seg * 16 + e) * PAD_ + kr]     = (__bf16)((float)x0[e] * w);
        Als[(seg * 16 + 8 + e) * PAD_ + kr] = (__bf16)((float)x1[e] * w);
      }
    }
    {   // B: transposed K, 32 n per thread
      const v8bf* p = (const v8bf*)(kmat + grow * MEM_ + seg * 32);
#pragma unroll
      for (int q4 = 0; q4 < 4; ++q4) {
        v8bf xx = p[q4];
#pragma unroll
        for (int e = 0; e < 8; ++e)
          Bls[(seg * 32 + q4 * 8 + e) * PAD_ + kr] = xx[e];
      }
    }
    __syncthreads();
    v16bf afrag[4];
#pragma unroll
    for (int mi = 0; mi < 4; ++mi)
      afrag[mi] = load_a_frag(Als + (wm * 64 + mi * 16) * PAD_, PAD_);
#pragma unroll
    for (int ni = 0; ni < 4; ++ni) {
      v16bf bfrag = load_b_frag(Bls + (wn * 64 + ni * 16) * PAD_, PAD_);
#pragma unroll
      for (int mi = 0; mi < 4; ++mi)
        acc[mi][ni] = WMMA_BF16(afrag[mi], bfrag, acc[mi][ni]);
    }
    __syncthreads();
  }

  float* Gc = G + chunk * (MEM_ * MEM_);
  const int crow = (lane >> 4) * 8;
  const int ccol = lane & 15;
#pragma unroll
  for (int mi = 0; mi < 4; ++mi)
#pragma unroll
    for (int ni = 0; ni < 4; ++ni)
#pragma unroll
      for (int v = 0; v < 8; ++v) {
        int gm = m0 + wm * 64 + mi * 16 + crow + v;
        int gn = wn * 64 + ni * 16 + ccol;
        Gc[gm * MEM_ + gn] = acc[mi][ni][v];
      }
}

// ---------- kernel: parallel-per-element sequential scan over chunks ----------
__global__ void fwm_state_scan(const float* __restrict__ G, const float* __restrict__ M0,
                               __bf16* __restrict__ states) {
  const int e = blockIdx.x * 256 + threadIdx.x;   // 0..65535
  float st = M0[e];
  const float dC = powf(DECAY_, (float)CHUNK_);
  for (int c = 0; c < NCHUNK_; ++c) {
    states[c * (MEM_ * MEM_) + e] = (__bf16)st;   // state at chunk start
    st = dC * st + G[c * (MEM_ * MEM_) + e];
  }
}

// ---------- kernel: per-chunk retrieval ----------
// r = mask ⊙ (Qs Ks^T) V / b  +  Qs @ Mstate^T   (Qs rows scaled decay^tau_i, Ks rows decay^(-1-tau_j))
// grid (NCHUNK_, 2 row-halves), block 256; block rows 128, cols 256
__global__ void __launch_bounds__(256) fwm_retrieve(
    const __bf16* __restrict__ q, const __bf16* __restrict__ kmat,
    const __bf16* __restrict__ vmat, const __bf16* __restrict__ states,
    __bf16* __restrict__ r) {
  __shared__ __align__(16) __bf16 Als[128 * PAD_];
  __shared__ __align__(16) __bf16 Bls[256 * PAD_];
  __shared__ __align__(16) __bf16 Sls[128 * SPAD_];
  const int tid  = threadIdx.x;
  const int lane = tid & 31;
  const int wid  = tid >> 5;
  const int wm   = wid >> 2;
  const int wn   = wid & 3;
  const int chunk = blockIdx.x;
  const int half  = blockIdx.y;
  const int row0  = chunk * 256 + half * 128;   // global first row of this block
  const int cbase = chunk * 256;

  const int srow  = tid >> 1;                   // 0..127 (block-local row)
  const int shalf = (tid & 1) * 16;
  const float wi  = powf(DECAY_, (float)((half * 128 + srow) >> 2));

  // ---- phase 1: scores S[i][j] over K = memory dim 256 ----
  v8f acc[4][4];
#pragma unroll
  for (int mi = 0; mi < 4; ++mi)
#pragma unroll
    for (int ni = 0; ni < 4; ++ni) acc[mi][ni] = vzero8();

  const int jrow = tid;                          // 0..255 (for B staging)
  const float wj = powf(DECAY_, -(float)(1 + (jrow >> 2)));

  for (int k0 = 0; k0 < 256; k0 += 32) {
    {   // A: Qs rows (scaled)
      const v8bf* pa = (const v8bf*)(q + (row0 + srow) * MEM_ + k0 + shalf);
      v8bf a0 = pa[0], a1 = pa[1];
      v8bf o0, o1;
#pragma unroll
      for (int e = 0; e < 8; ++e) {
        o0[e] = (__bf16)((float)a0[e] * wi);
        o1[e] = (__bf16)((float)a1[e] * wi);
      }
      *(v8bf*)(Als + srow * PAD_ + shalf)     = o0;
      *(v8bf*)(Als + srow * PAD_ + shalf + 8) = o1;
    }
    {   // B: K rows (scaled), natural row-major == Bt layout
      const v8bf* pb = (const v8bf*)(kmat + (cbase + jrow) * MEM_ + k0);
#pragma unroll
      for (int q4 = 0; q4 < 4; ++q4) {
        v8bf xx = pb[q4];
        v8bf oo;
#pragma unroll
        for (int e = 0; e < 8; ++e) oo[e] = (__bf16)((float)xx[e] * wj);
        *(v8bf*)(Bls + jrow * PAD_ + q4 * 8) = oo;
      }
    }
    __syncthreads();
    v16bf afrag[4];
#pragma unroll
    for (int mi = 0; mi < 4; ++mi)
      afrag[mi] = load_a_frag(Als + (wm * 64 + mi * 16) * PAD_, PAD_);
#pragma unroll
    for (int ni = 0; ni < 4; ++ni) {
      v16bf bfrag = load_b_frag(Bls + (wn * 64 + ni * 16) * PAD_, PAD_);
#pragma unroll
      for (int mi = 0; mi < 4; ++mi)
        acc[mi][ni] = WMMA_BF16(afrag[mi], bfrag, acc[mi][ni]);
    }
    __syncthreads();
  }

  // ---- mask (tau_j < tau_i), /b, dump bf16 scores to LDS ----
  const int crow = (lane >> 4) * 8;
  const int ccol = lane & 15;
#pragma unroll
  for (int mi = 0; mi < 4; ++mi)
#pragma unroll
    for (int ni = 0; ni < 4; ++ni)
#pragma unroll
      for (int v = 0; v < 8; ++v) {
        int il = wm * 64 + mi * 16 + crow + v;   // 0..127
        int jj = wn * 64 + ni * 16 + ccol;       // 0..255
        int ti = (half * 128 + il) >> 2;
        int tj = jj >> 2;
        float val = (tj < ti) ? acc[mi][ni][v] * 0.25f : 0.0f;
        Sls[il * SPAD_ + jj] = (__bf16)val;
      }
  __syncthreads();

  // ---- phase 2: r_intra = Smask @ V  (K over j = 256) ----
  v8f acc2[4][4];
#pragma unroll
  for (int mi = 0; mi < 4; ++mi)
#pragma unroll
    for (int ni = 0; ni < 4; ++ni) acc2[mi][ni] = vzero8();

  const int kr  = tid >> 3;   // 0..31
  const int seg = tid & 7;    // 0..7
  for (int k0 = 0; k0 < 256; k0 += 32) {
    {   // stage V^T: Bls[m][jr]
      const v8bf* p = (const v8bf*)(vmat + (cbase + k0 + kr) * MEM_ + seg * 32);
#pragma unroll
      for (int q4 = 0; q4 < 4; ++q4) {
        v8bf xx = p[q4];
#pragma unroll
        for (int e = 0; e < 8; ++e)
          Bls[(seg * 32 + q4 * 8 + e) * PAD_ + kr] = xx[e];
      }
    }
    __syncthreads();
    v16bf afrag[4];
#pragma unroll
    for (int mi = 0; mi < 4; ++mi)
      afrag[mi] = load_a_frag(Sls + (wm * 64 + mi * 16) * SPAD_ + k0, SPAD_);
#pragma unroll
    for (int ni = 0; ni < 4; ++ni) {
      v16bf bfrag = load_b_frag(Bls + (wn * 64 + ni * 16) * PAD_, PAD_);
#pragma unroll
      for (int mi = 0; mi < 4; ++mi)
        acc2[mi][ni] = WMMA_BF16(afrag[mi], bfrag, acc2[mi][ni]);
    }
    __syncthreads();
  }

  // ---- phase 3: r_inter += Qs @ Mstate^T (K over feature n = 256) ----
  const __bf16* st = states + chunk * (MEM_ * MEM_);
  for (int k0 = 0; k0 < 256; k0 += 32) {
    {   // A: Qs restage
      const v8bf* pa = (const v8bf*)(q + (row0 + srow) * MEM_ + k0 + shalf);
      v8bf a0 = pa[0], a1 = pa[1];
      v8bf o0, o1;
#pragma unroll
      for (int e = 0; e < 8; ++e) {
        o0[e] = (__bf16)((float)a0[e] * wi);
        o1[e] = (__bf16)((float)a1[e] * wi);
      }
      *(v8bf*)(Als + srow * PAD_ + shalf)     = o0;
      *(v8bf*)(Als + srow * PAD_ + shalf + 8) = o1;
    }
    {   // B: state rows, natural row-major == Bt layout; thread -> m = tid
#if FWM_ASYNC
      const __bf16* ps = st + tid * MEM_ + k0;
      __bf16* lp = Bls + tid * PAD_;
      async_cp16(ps,      lp);
      async_cp16(ps + 8,  lp + 8);
      async_cp16(ps + 16, lp + 16);
      async_cp16(ps + 24, lp + 24);
      FWM_WAIT_ASYNC();
#else
      const v8bf* ps = (const v8bf*)(st + tid * MEM_ + k0);
      v8bf s0 = ps[0], s1 = ps[1], s2 = ps[2], s3 = ps[3];
      *(v8bf*)(Bls + tid * PAD_ + 0)  = s0;
      *(v8bf*)(Bls + tid * PAD_ + 8)  = s1;
      *(v8bf*)(Bls + tid * PAD_ + 16) = s2;
      *(v8bf*)(Bls + tid * PAD_ + 24) = s3;
#endif
    }
    __syncthreads();
    v16bf afrag[4];
#pragma unroll
    for (int mi = 0; mi < 4; ++mi)
      afrag[mi] = load_a_frag(Als + (wm * 64 + mi * 16) * PAD_, PAD_);
#pragma unroll
    for (int ni = 0; ni < 4; ++ni) {
      v16bf bfrag = load_b_frag(Bls + (wn * 64 + ni * 16) * PAD_, PAD_);
#pragma unroll
      for (int mi = 0; mi < 4; ++mi)
        acc2[mi][ni] = WMMA_BF16(afrag[mi], bfrag, acc2[mi][ni]);
    }
    __syncthreads();
  }

  // ---- epilogue: r bf16 ----
#pragma unroll
  for (int mi = 0; mi < 4; ++mi)
#pragma unroll
    for (int ni = 0; ni < 4; ++ni)
#pragma unroll
      for (int v = 0; v < 8; ++v) {
        int il  = wm * 64 + mi * 16 + crow + v;
        int col = wn * 64 + ni * 16 + ccol;
        r[(row0 + il) * MEM_ + col] = (__bf16)acc2[mi][ni][v];
      }
}

__global__ void fwm_write_gate(const float* __restrict__ gate, float* __restrict__ o) {
  o[0] = 1.0f / (1.0f + expf(-gate[0]));
}

// ---------- host launcher ----------
extern "C" void kernel_launch(void* const* d_in, const int* in_sizes, int n_in,
                              void* d_out, int out_size, void* d_ws, size_t ws_size,
                              hipStream_t stream) {
  const float* x    = (const float*)d_in[0];
  const float* Wq   = (const float*)d_in[1];
  const float* Wk   = (const float*)d_in[2];
  const float* Wv   = (const float*)d_in[3];
  const float* Wo   = (const float*)d_in[4];
  const float* gate = (const float*)d_in[5];
  const float* M0   = (const float*)d_in[6];
  float* out = (float*)d_out;

  char* ws = (char*)d_ws;
  // region A: xr (bf16, 32 MB) is dead after QKV; reuse for G + states + rbf
  __bf16* xr     = (__bf16*)(ws + 0);
  float*  G      = (float*) (ws + 0);                  // 64*65536*4  = 16 MB
  __bf16* states = (__bf16*)(ws + 16777216);           // 64*65536*2  =  8 MB
  __bf16* rbf    = (__bf16*)(ws + 25165824);           // 16384*256*2 =  8 MB
  __bf16* wqb    = (__bf16*)(ws + 33554432);
  __bf16* wkb    = (__bf16*)(ws + 34078720);
  __bf16* wvb    = (__bf16*)(ws + 34603008);
  __bf16* wob    = (__bf16*)(ws + 35127296);
  __bf16* qb     = (__bf16*)(ws + 35651584);
  __bf16* kb     = (__bf16*)(ws + 44040192);
  __bf16* vb     = (__bf16*)(ws + 52428800);

  // 1) convert inputs to bf16 (x reordered to time-major rows)
  fwm_convert_x<<<8192, 256, 0, stream>>>(x, xr);
  fwm_convert_w<<<128, 256, 0, stream>>>(Wq, wqb);
  fwm_convert_w<<<128, 256, 0, stream>>>(Wk, wkb);
  fwm_convert_w<<<128, 256, 0, stream>>>(Wv, wvb);
  fwm_convert_w<<<128, 256, 0, stream>>>(Wo, wob);

  // 2) QKV projections: (16384x1024) x (256x1024)^T
  dim3 gqkv(ROWS_ / 128, MEM_ / 128);
  fwm_gemm<<<gqkv, 256, 0, stream>>>(xr, H_, wqb, H_, H_, qb, MEM_, nullptr, nullptr);
  fwm_gemm<<<gqkv, 256, 0, stream>>>(xr, H_, wkb, H_, H_, kb, MEM_, nullptr, nullptr);
  fwm_gemm<<<gqkv, 256, 0, stream>>>(xr, H_, wvb, H_, H_, vb, MEM_, nullptr, nullptr);

  // 3) per-chunk decay-weighted outer-product sums (overwrites xr region)
  fwm_chunk_sums<<<dim3(NCHUNK_, 2), 256, 0, stream>>>(vb, kb, G);

  // 4) sequential chunk recurrence, parallel across the 65536 state elements
  fwm_state_scan<<<256, 256, 0, stream>>>(G, M0, states);

  // 5) per-chunk retrieval (intra flash-style + inter vs. carried state)
  fwm_retrieve<<<dim3(NCHUNK_, 2), 256, 0, stream>>>(qb, kb, vb, states, rbf);

  // 6) output projection + gate + scatter to (b,s,h)
  fwm_gemm<<<dim3(ROWS_ / 128, H_ / 128), 256, 0, stream>>>(
      rbf, MEM_, wob, MEM_, MEM_, nullptr, 0, out, gate);

  // 7) gate_value scalar
  fwm_write_gate<<<1, 1, 0, stream>>>(gate, out + (size_t)ROWS_ * H_);
}
